// TestLSTM_24292335026485
// MI455X (gfx1250) — matrix operand order repaired and checked by reading, not compiled, not run on hardware
//
#include <hip/hip_runtime.h>
#include <hip/hip_bf16.h>
#include <stdint.h>
#include <stddef.h>

// Problem sizes (fixed by the reference)
#define B_  64
#define T_  1024
#define I_  256
#define H_  512

typedef _Float16 half_t;
typedef __attribute__((ext_vector_type(16))) _Float16 v16h;
typedef __attribute__((ext_vector_type(8)))  _Float16 v8h;
typedef __attribute__((ext_vector_type(8)))  float    v8f;

// ---------------- workspace layout (bytes, all offsets 256-aligned) ----------
//  xt16      [T][B][I]  f16                          : 33,554,432
//  wih_frag  [128 nT][8 kT][32 lane][8 dw] (u32)     :  1,048,576
//  whh_frag  [128 nT][16 kT][32 lane][8 dw] (u32)    :  2,097,152
//  bias      [4H] f32                                :      8,192
//  h16       [B][H] f16                              :     65,536
//  c         [B][H] f32                              :    131,072
static const size_t XT_OFF   = 0;
static const size_t WIH_OFF  = 33554432;
static const size_t WHH_OFF  = WIH_OFF + 1048576;      // 34603008
static const size_t BIAS_OFF = WHH_OFF + 2097152;      // 36700160
static const size_t H16_OFF  = BIAS_OFF + 8192;        // 36708352
static const size_t C_OFF    = H16_OFF + 65536;        // 36773888

// ---------------------------------------------------------------------------
// x [B][T][I] f32  ->  xt16 [T][B][I] f16  (time-major so per-step A tiles are
// contiguous per batch row)
__global__ void k_convert_x(const float* __restrict__ x, half_t* __restrict__ xt)
{
    size_t idx = (size_t)blockIdx.x * blockDim.x + threadIdx.x;   // over T*B*I
    if (idx >= (size_t)T_ * B_ * I_) return;
    int i = (int)(idx % I_);
    int b = (int)((idx / I_) % B_);
    int t = (int)(idx / ((size_t)I_ * B_));
    xt[idx] = (half_t)x[(size_t)b * T_ * I_ + (size_t)t * I_ + i];
}

// Pre-swizzle a weight matrix w [4H][K] f32 (row-major) into per-wave WMMA
// B-fragments (B = w^T, shape K x 4H).  ISA 16-bit B layout for 16x16x32:
//   lanes 0-15 : N = nTile*16 + lane,      K = kTile*32 + {0..7,16..23}
//   lanes 16-31: N = nTile*16 + lane - 16, K = kTile*32 + {8..15,24..31}
// Fragment f = nTile*kTiles + kTile; dword d = f*256 + lane*8 + v.
__global__ void k_pack_w(const float* __restrict__ w, uint32_t* __restrict__ wf,
                         int K, int kTiles)
{
    int d = blockIdx.x * blockDim.x + threadIdx.x;
    int total = 128 * kTiles * 256;
    if (d >= total) return;
    int frag  = d >> 8;
    int rem   = d & 255;
    int lane  = rem >> 3;
    int v     = rem & 7;
    int nTile = frag / kTiles;
    int kTile = frag % kTiles;
    int n = nTile * 16 + (lane & 15);
    int k = kTile * 32 + ((lane >> 4) << 3) + 2 * v + ((v >> 2) << 3);
    union { _Float16 h[2]; uint32_t u; } p;
    p.h[0] = (_Float16)w[(size_t)n * K + k];
    p.h[1] = (_Float16)w[(size_t)n * K + k + 1];
    wf[d] = p.u;
}

// bias = b_ih + b_hh ; zero h16 and c state
__global__ void k_init(const float* __restrict__ b_ih, const float* __restrict__ b_hh,
                       float* __restrict__ bias, half_t* __restrict__ h16,
                       float* __restrict__ c)
{
    int i = blockIdx.x * blockDim.x + threadIdx.x;   // 32768 threads
    if (i < 4 * H_) bias[i] = b_ih[i] + b_hh[i];
    if (i < B_ * H_) { h16[i] = (half_t)0.0f; c[i] = 0.0f; }
}

// ---------------------------------------------------------------------------
// A-fragment loader for 16-bit A 16x32 (M x K), row-major source with
// row = lane&15; lane<16 takes K {0..7,16..23}, lane>=16 takes K {8..15,24..31}.
__device__ __forceinline__ v16h load_a_frag(const half_t* __restrict__ rowptr, int koff)
{
    v8h lo = *(const v8h*)(rowptr + koff);
    v8h hi = *(const v8h*)(rowptr + koff + 16);
    return __builtin_shufflevector(lo, hi, 0,1,2,3,4,5,6,7,8,9,10,11,12,13,14,15);
}

__device__ __forceinline__ float sigmoidf_(float x)
{
    return 1.0f / (1.0f + __expf(-x));
}

// One LSTM timestep, fully fused.  grid = (32 hTiles, 4 bTiles), block = 256
// threads = 8 waves.  Each workgroup owns one 16(batch) x 16(h) output tile;
// the combined K space (I + H = 768, i.e. 24 k-steps of 32) is split across
// 8 waves as 4 gates x 2 K-halves, so each wave runs a short 12-WMMA chain
// with a single v8f accumulator (low pressure -> deep load prefetch).
// Partials are combined via LDS; wave 0 does the bias + sigmoid/tanh + cell
// update epilogue.
__global__ __launch_bounds__(256) void k_step(
    const half_t*  __restrict__ xt,     // [T][B][I] f16
    const uint32_t* __restrict__ wihf,  // B-fragments of w_ih^T
    const uint32_t* __restrict__ whhf,  // B-fragments of w_hh^T
    const float*   __restrict__ bias,   // [4H]
    half_t*        __restrict__ h16,    // [B][H] f16 state (in/out)
    float*         __restrict__ c,      // [B][H] f32 cell state (in/out)
    float*         __restrict__ out,    // [B][T][H]
    float*         __restrict__ hn,     // [B][H]
    float*         __restrict__ cn,     // [B][H]
    int t)
{
    __shared__ __align__(32) float lds[8 * 256];

    const int lane  = threadIdx.x & 31;
    const int wave  = threadIdx.x >> 5;   // 0..7
    const int g     = wave & 3;           // gate: i, f, g, o
    const int khalf = wave >> 2;          // 0: x-proj + h[0:128);  1: h[128:512)
    const int th    = blockIdx.x;         // h tile   0..31
    const int tb    = blockIdx.y;         // b tile   0..3
    const int rowA  = tb * 16 + (lane & 15);
    const int kbase = (lane >> 4) << 3;   // 0 or 8 (A-layout half select)

    v8f acc = {};

    const half_t* xrow = xt + (size_t)t * B_ * I_ + (size_t)rowA * I_;
    const half_t* hrow = h16 + (size_t)rowA * H_;
    const uint32_t* wib = wihf + (size_t)lane * 8 + (size_t)(g * 32 + th) * (8  * 256);
    const uint32_t* whb = whhf + (size_t)lane * 8 + (size_t)(g * 32 + th) * (16 * 256);

    if (khalf == 0) {
        // input projection: K = I (8 k-steps)
        #pragma unroll
        for (int kk = 0; kk < 8; ++kk) {
            v16h a = load_a_frag(xrow, kk * 32 + kbase);
            v16h b = *(const v16h*)(wib + (size_t)kk * 256);
            acc = __builtin_amdgcn_wmma_f32_16x16x32_f16(false, a, false, b, (short)0, acc, false, false);
        }
        // recurrent projection, first quarter: h K in [0,128)
        #pragma unroll
        for (int kk = 0; kk < 4; ++kk) {
            v16h a = load_a_frag(hrow, kk * 32 + kbase);
            v16h b = *(const v16h*)(whb + (size_t)kk * 256);
            acc = __builtin_amdgcn_wmma_f32_16x16x32_f16(false, a, false, b, (short)0, acc, false, false);
        }
    } else {
        // recurrent projection, remaining: h K in [128,512)
        #pragma unroll
        for (int kk = 4; kk < 16; ++kk) {
            v16h a = load_a_frag(hrow, kk * 32 + kbase);
            v16h b = *(const v16h*)(whb + (size_t)kk * 256);
            acc = __builtin_amdgcn_wmma_f32_16x16x32_f16(false, a, false, b, (short)0, acc, false, false);
        }
    }

    // publish partials
    *(v8f*)&lds[wave * 256 + lane * 8] = acc;
    __syncthreads();

    // wave 0: combine K-halves, add bias, nonlinearities, state update.
    // C/D layout: col N = lane&15, row M = 8*(lane>=16) + j.
    if (threadIdx.x < 32) {
        const int e0 = lane * 8;
        v8f pi0 = *(const v8f*)&lds[0 * 256 + e0];
        v8f pf0 = *(const v8f*)&lds[1 * 256 + e0];
        v8f pg0 = *(const v8f*)&lds[2 * 256 + e0];
        v8f po0 = *(const v8f*)&lds[3 * 256 + e0];
        v8f pi1 = *(const v8f*)&lds[4 * 256 + e0];
        v8f pf1 = *(const v8f*)&lds[5 * 256 + e0];
        v8f pg1 = *(const v8f*)&lds[6 * 256 + e0];
        v8f po1 = *(const v8f*)&lds[7 * 256 + e0];

        const int hcol = th * 16 + (lane & 15);
        const float bi = bias[0 * H_ + hcol];
        const float bf = bias[1 * H_ + hcol];
        const float bg = bias[2 * H_ + hcol];
        const float bo = bias[3 * H_ + hcol];
        const int rbase = tb * 16 + ((lane >> 4) << 3);
        const bool last = (t == T_ - 1);

        #pragma unroll
        for (int j = 0; j < 8; ++j) {
            int brow = rbase + j;
            float iv = sigmoidf_(pi0[j] + pi1[j] + bi);
            float fv = sigmoidf_(pf0[j] + pf1[j] + bf);
            float gv = tanhf(pg0[j] + pg1[j] + bg);
            float ov = sigmoidf_(po0[j] + po1[j] + bo);
            int ci = brow * H_ + hcol;
            float cp  = c[ci];
            float cnw = fv * cp + iv * gv;
            float hv  = ov * tanhf(cnw);
            c[ci]   = cnw;
            h16[ci] = (half_t)hv;
            out[(size_t)brow * T_ * H_ + (size_t)t * H_ + hcol] = hv;
            if (last) { hn[ci] = hv; cn[ci] = cnw; }
        }
    }
}

// ---------------------------------------------------------------------------
extern "C" void kernel_launch(void* const* d_in, const int* in_sizes, int n_in,
                              void* d_out, int out_size, void* d_ws, size_t ws_size,
                              hipStream_t stream)
{
    (void)in_sizes; (void)n_in; (void)out_size; (void)ws_size;

    const float* x    = (const float*)d_in[0];   // [B,T,I]
    const float* w_ih = (const float*)d_in[1];   // [4H,I]
    const float* w_hh = (const float*)d_in[2];   // [4H,H]
    const float* b_ih = (const float*)d_in[3];   // [4H]
    const float* b_hh = (const float*)d_in[4];   // [4H]

    float* out = (float*)d_out;                      // [B,T,H]
    float* hn  = out + (size_t)B_ * T_ * H_;         // [1,B,H]
    float* cn  = hn + (size_t)B_ * H_;               // [1,B,H]

    char* ws = (char*)d_ws;
    half_t*   xt   = (half_t*)(ws + XT_OFF);
    uint32_t* wihf = (uint32_t*)(ws + WIH_OFF);
    uint32_t* whhf = (uint32_t*)(ws + WHH_OFF);
    float*    bias = (float*)(ws + BIAS_OFF);
    half_t*   h16  = (half_t*)(ws + H16_OFF);
    float*    c    = (float*)(ws + C_OFF);

    // prep
    {
        size_t n = (size_t)T_ * B_ * I_;
        k_convert_x<<<(unsigned)((n + 255) / 256), 256, 0, stream>>>(x, xt);
    }
    k_pack_w<<<(128 * 8  * 256) / 256, 256, 0, stream>>>(w_ih, wihf, I_, 8);
    k_pack_w<<<(128 * 16 * 256) / 256, 256, 0, stream>>>(w_hh, whhf, H_, 16);
    k_init<<<(B_ * H_) / 256, 256, 0, stream>>>(b_ih, b_hh, bias, h16, c);

    // recurrent scan: one fused kernel per timestep (graph-friendly, deterministic)
    dim3 grid(32, 4);
    for (int t = 0; t < T_; ++t) {
        k_step<<<grid, 256, 0, stream>>>(xt, wihf, whhf, bias, h16, c, out, hn, cn, t);
    }
}
